// FishnetGCN_62096637165586
// MI455X (gfx1250) — compile-verified, matrix operand order checked by reading, not run on hardware
//
#include <hip/hip_runtime.h>
#include <hip/hip_bf16.h>
#include <math.h>

typedef __attribute__((ext_vector_type(16))) _Float16 v16h;
typedef __attribute__((ext_vector_type(8)))  float    v8f;
typedef unsigned int u32x4 __attribute__((ext_vector_type(4)));
typedef int          i32x8 __attribute__((ext_vector_type(8)));
typedef int          i32x4 __attribute__((ext_vector_type(4)));

#define N_NODES 20000
#define N_EDGES 320000
#define HID     128
#define NP      16
#define FDIM    152   /* 16 + 136 */
#define TRI     136

// ---------------------------------------------------------------------------
// TDM: 1-D tile copy global -> LDS (wave-level DMA, tracked by TENSORcnt)
// ---------------------------------------------------------------------------
__device__ __forceinline__ void tdm_load_1d(void* ldsPtr, const void* gptr,
                                            unsigned nElems) {
  unsigned lds = (unsigned)(size_t)ldsPtr;       // LDS byte offset (addr[31:0])
  unsigned long long ga = (unsigned long long)(size_t)gptr;
  u32x4 g0;
  g0[0] = 1u;                                     // count=1 (valid descriptor)
  g0[1] = lds;                                    // lds_addr
  g0[2] = (unsigned)ga;                           // global_addr[31:0]
  g0[3] = (unsigned)((ga >> 32) & 0x01ffffffu) | (2u << 30);  // addr[56:32]|type=2
  i32x8 g1;
  g1[0] = (int)(2u << 16);                        // data_size=2 -> 4 bytes
  g1[1] = (int)((nElems & 0xffffu) << 16);        // tensor_dim0 lo16
  g1[2] = (int)(((nElems >> 16) & 0xffffu) | (1u << 16)); // dim0 hi | tensor_dim1=1
  g1[3] = (int)((nElems & 0xffffu) << 16);        // tile_dim0
  g1[4] = 1;                                      // tile_dim1=1
  g1[5] = (int)nElems;                            // tensor_dim0_stride lo32
  g1[6] = 0;
  g1[7] = 0;
  i32x4 gz;
  gz[0] = gz[1] = gz[2] = gz[3] = 0;
  i32x8 gz8;
  gz8[0] = gz8[1] = gz8[2] = gz8[3] = 0;
  gz8[4] = gz8[5] = gz8[6] = gz8[7] = 0;
  __builtin_amdgcn_tensor_load_to_lds(g0, g1, gz, gz, gz8, 0);
}

// ---------------------------------------------------------------------------
// Weight pre-swizzle: f32 [layers][K][N] -> f16 WMMA-B fragments
// out layout: [layer][kt][nt][lane][16] halves, 32B contiguous per fragment
// ---------------------------------------------------------------------------
__global__ void fishnet_swz_kernel(const float* __restrict__ w,
                                   _Float16* __restrict__ out, int K, int N,
                                   int Npad, int layers) {
  int KT = K >> 5, NT = Npad >> 4;
  int total = layers * KT * NT * 512;
  int idx = blockIdx.x * blockDim.x + threadIdx.x;
  if (idx >= total) return;
  int j = idx & 15;
  int lane = (idx >> 4) & 31;
  int rest = idx >> 9;
  int nt = rest % NT;
  int kt = (rest / NT) % KT;
  int layer = rest / (NT * KT);
  int kb = (lane >> 4) * 8;
  int n = lane & 15;
  int k = kt * 32 + ((j < 8) ? (kb + j) : (16 + kb + (j - 8)));
  int col = nt * 16 + n;
  float v = (col < N) ? w[((size_t)layer * K + k) * N + col] : 0.f;
  out[idx] = (_Float16)v;
}

// ---------------------------------------------------------------------------
// Node encoder: h = x @ Wn[8,128] + bn
// ---------------------------------------------------------------------------
__global__ void fishnet_enc_kernel(const float* __restrict__ in,
                                   const float* __restrict__ W,
                                   const float* __restrict__ b,
                                   float* __restrict__ out, int nRows) {
  int idx = blockIdx.x * blockDim.x + threadIdx.x;
  int n = idx >> 7, c = idx & 127;
  if (n >= nRows) return;
  float acc = b[c];
#pragma unroll
  for (int q = 0; q < 8; ++q) acc += in[n * 8 + q] * W[q * 128 + c];
  out[idx] = acc;
}

// ---------------------------------------------------------------------------
// LayerNorm (+optional relu), one node per 128-thread block
// ---------------------------------------------------------------------------
__global__ __launch_bounds__(128) void fishnet_ln_kernel(
    const float* __restrict__ h, const float* __restrict__ g,
    const float* __restrict__ b, float* __restrict__ out, int nNodes,
    int doRelu) {
  __shared__ float red[128];
  int n = blockIdx.x;
  int tid = threadIdx.x;
  if (n >= nNodes) return;
  float v = h[n * 128 + tid];
  red[tid] = v;
  __syncthreads();
  for (int s = 64; s > 0; s >>= 1) {
    if (tid < s) red[tid] += red[tid + s];
    __syncthreads();
  }
  float mu = red[0] / 128.f;
  __syncthreads();
  float d = v - mu;
  red[tid] = d * d;
  __syncthreads();
  for (int s = 64; s > 0; s >>= 1) {
    if (tid < s) red[tid] += red[tid + s];
    __syncthreads();
  }
  float rs = rsqrtf(red[0] / 128.f + 1e-5f);
  float o = d * rs * g[tid] + b[tid];
  if (doRelu) o = o > 0.f ? o : 0.f;
  out[n * 128 + tid] = o;
}

__global__ void fishnet_zero_kernel(float4* __restrict__ p, int n4) {
  int i = blockIdx.x * blockDim.x + threadIdx.x;
  if (i < n4) p[i] = make_float4(0.f, 0.f, 0.f, 0.f);
}

// ---------------------------------------------------------------------------
// Edge kernel: 64 edges per block (4 waves x 16-edge WMMA tiles)
// TDM stages edge_attr/ids/encoder weights into LDS; msg built + stored in
// A-fragment-swizzled f16 LDS; B from pre-swizzled f16 weights (b128 loads);
// score from accumulators -> atomic s_n; Fe = L*L^T via wmma(a,a) -> atomic F_n
// ---------------------------------------------------------------------------
__global__ __launch_bounds__(128) void fishnet_edge_kernel(
    const float* __restrict__ hin, const float* __restrict__ eattr,
    const int* __restrict__ srcIdx, const int* __restrict__ dstIdx,
    const float* __restrict__ We, const float* __restrict__ be,
    const _Float16* __restrict__ a1wz, const float* __restrict__ a1b,
    float* __restrict__ s_n, float* __restrict__ F_n) {
  __shared__ __attribute__((aligned(32))) float    sWe[8 * 128];
  __shared__ __attribute__((aligned(32))) float    sbe[128];
  __shared__ __attribute__((aligned(32))) float    sea[64 * 8];
  __shared__ __attribute__((aligned(32))) int      sdst[64];
  __shared__ __attribute__((aligned(32))) int      ssrc[64];
  __shared__ __attribute__((aligned(32))) _Float16 sA[4][4][32][16]; // A frags
  __shared__ __attribute__((aligned(32))) float    sy[4][16][144];   // Q cols
  __shared__ __attribute__((aligned(32))) _Float16 sL[4][32][16];    // L frags

  const int tid = threadIdx.x;
  const int lane = tid & 31, wave = tid >> 5;
  const int e0 = blockIdx.x * 64;

  if (wave == 0) {                     // wave-level DMA copies
    tdm_load_1d(sWe, We, 8 * 128);
    tdm_load_1d(sbe, be, 128);
    tdm_load_1d(sea, eattr + (size_t)e0 * 8, 64 * 8);
    tdm_load_1d(sdst, dstIdx + e0, 64);
    tdm_load_1d(ssrc, srcIdx + e0, 64);
    __builtin_amdgcn_s_wait_tensorcnt(0);
  }
  __syncthreads();

  // ---- build msg = relu(hin[src] + ea@We + be) + 1e-7, store A-swizzled ----
  for (int it = 0; it < 64; ++it) {
    int s = ssrc[it];
    float ea = sbe[tid];
#pragma unroll
    for (int q = 0; q < 8; ++q) ea += sea[it * 8 + q] * sWe[q * 128 + tid];
    float v = hin[(size_t)s * 128 + tid] + ea;
    v = (v > 0.f ? v : 0.f) + 1e-7f;
    int w = it >> 4, mrow = it & 15;
    int k = tid & 31, kt = tid >> 5;
    sA[w][kt][mrow + (((k >> 3) & 1) << 4)][(((k >> 4) & 1) << 3) | (k & 7)] =
        (_Float16)v;
  }
  __syncthreads();

  // ---- GEMM: y = msg @ a1w + a1b (10 N-tiles x 4 K-steps) ----
  v16h afr[4];
#pragma unroll
  for (int kt = 0; kt < 4; ++kt)
    afr[kt] = *(const v16h*)&sA[wave][kt][lane][0];

  const int n = lane & 15;
  for (int nt = 0; nt < 10; ++nt) {
    int col = nt * 16 + n;
    float bias = (col < FDIM) ? a1b[col] : 0.f;
    v8f c;
#pragma unroll
    for (int i = 0; i < 8; ++i) c[i] = bias;
#pragma unroll
    for (int kt = 0; kt < 4; ++kt) {
      v16h bfr = *(const v16h*)(a1wz + ((((kt * 10) + nt) * 32 + lane) << 4));
      c = __builtin_amdgcn_wmma_f32_16x16x32_f16(false, afr[kt], false, bfr,
                                                 (short)0, c, false, false);
    }
    if (nt == 0) {
#pragma unroll
      for (int i = 0; i < 8; ++i) {
        int r = i + ((lane >> 4) << 3);
        atomicAdd(&s_n[(size_t)sdst[wave * 16 + r] * NP + n], c[i]);
      }
    } else {
#pragma unroll
      for (int i = 0; i < 8; ++i) {
        int r = i + ((lane >> 4) << 3);
        int t = nt * 16 - 16 + n;
        if (t < TRI) sy[wave][r][t] = c[i];
      }
    }
  }
  __builtin_amdgcn_wave_barrier();

  // zero the K>=16 halves of the L fragment once
#pragma unroll
  for (int i = 8; i < 16; ++i) sL[wave][lane][i] = (_Float16)0.f;

  // ---- per edge: L from Q (FILL closed form); softplus only on diagonal ----
  for (int r = 0; r < 16; ++r) {
#pragma unroll
    for (int u = 0; u < 8; ++u) {
      int p = lane * 8 + u;                       // flattened 16x16 index
      int qi = p >> 4, qj = p & 15;
      int t = (p < 120) ? (p + 16) : (255 - p);   // FILL_IDX closed form
      sL[wave][qi + (((qj >> 3) & 1) << 4)][qj & 7] = (_Float16)sy[wave][r][t];
    }
    if (lane < 16) {                              // 16 diag entries only
      int p = 17 * lane;
      int t = (p < 120) ? (p + 16) : (255 - p);
      float q = sy[wave][r][t];
      float sp = (q > 20.f) ? q : log1pf(__expf(q));
      sL[wave][lane + (((lane >> 3) & 1) << 4)][lane & 7] = (_Float16)sp;
    }
    __builtin_amdgcn_wave_barrier();
    v16h lf = *(const v16h*)&sL[wave][lane][0];
    v8f fc;
#pragma unroll
    for (int i = 0; i < 8; ++i) fc[i] = 0.f;
    fc = __builtin_amdgcn_wmma_f32_16x16x32_f16(false, lf, false, lf, (short)0,
                                                fc, false, false);
    int d = sdst[wave * 16 + r];
    // one base address, 8 atomics with immediate offsets
    float* fp = F_n + (size_t)d * (NP * NP) + ((lane >> 4) << 7) + (lane & 15);
#pragma unroll
    for (int i = 0; i < 8; ++i) atomicAdd(fp + i * NP, fc[i]);
    __builtin_amdgcn_wave_barrier();
  }
}

// ---------------------------------------------------------------------------
// Solve: (F_n + I) mle = s_n, Gauss-Jordan rows-in-lanes (triangular updates);
// a2w/a2b TDM-staged to LDS; mid = mle@a2w + a2b + xin
// ---------------------------------------------------------------------------
__global__ __launch_bounds__(128) void fishnet_solve_kernel(
    const float* __restrict__ F_n, const float* __restrict__ s_n,
    const float* __restrict__ a2w, const float* __restrict__ a2b,
    const float* __restrict__ xin, float* __restrict__ mid, int nNodes) {
  __shared__ __attribute__((aligned(32))) float sW[16 * 128];
  __shared__ __attribute__((aligned(32))) float sb[128];
  __shared__ float sMle[4][16];
  int tid = threadIdx.x, lane = tid & 31, wave = tid >> 5;
  int node = blockIdx.x * 4 + wave;

  if (wave == 0) {
    tdm_load_1d(sW, a2w, 16 * 128);
    tdm_load_1d(sb, a2b, 128);
    __builtin_amdgcn_s_wait_tensorcnt(0);
  }
  __syncthreads();

  int r = lane & 15;
  float row[17];
#pragma unroll
  for (int j = 0; j < 16; ++j)
    row[j] = F_n[(size_t)node * 256 + r * 16 + j] + (j == r ? 1.f : 0.f);
  row[16] = s_n[(size_t)node * 16 + r];

#pragma unroll
  for (int k = 0; k < 16; ++k) {
    float pivkk = __shfl(row[k], k, 32);
    float f = row[k] / pivkk;
    bool isPivot = (r == k);
#pragma unroll
    for (int j = k; j < 17; ++j) {    // cols < k already eliminated
      float pj = __shfl(row[j], k, 32);
      row[j] = isPivot ? (row[j] / pivkk) : (row[j] - f * pj);
    }
  }
  if (lane < 16) sMle[wave][r] = row[16];
  __builtin_amdgcn_wave_barrier();

  float mle[16];
#pragma unroll
  for (int k = 0; k < 16; ++k) mle[k] = sMle[wave][k];
#pragma unroll
  for (int jj = 0; jj < 4; ++jj) {
    int col = lane + jj * 32;
    float acc = sb[col];
#pragma unroll
    for (int k = 0; k < 16; ++k) acc += mle[k] * sW[k * 128 + col];
    mid[(size_t)node * 128 + col] = acc + xin[(size_t)node * 128 + col];
  }
}

// ---------------------------------------------------------------------------
// Fused node MLP: hout = [h +] relu(LN(mid@m1w + m1b))@m2w + m2b
// ---------------------------------------------------------------------------
__global__ __launch_bounds__(128) void fishnet_mlp_kernel(
    const float* __restrict__ mid, const float* __restrict__ hprev,
    const _Float16* __restrict__ m1wz, const float* __restrict__ m1b,
    const float* __restrict__ lng, const float* __restrict__ lnb,
    const _Float16* __restrict__ m2wz, const float* __restrict__ m2b,
    float* __restrict__ hout, int residual, int nNodes) {
  __shared__ __attribute__((aligned(32))) _Float16 sA[4][32][16];
  __shared__ __attribute__((aligned(32))) float    sT[16][264];
  __shared__ __attribute__((aligned(32))) _Float16 sB[8][32][16];
  __shared__ float sMu[16], sRs[16];

  int tid = threadIdx.x, lane = tid & 31, wave = tid >> 5;
  int n0 = blockIdx.x * 16;

  for (int i = tid; i < 16 * 128; i += 128) {
    int rr = i >> 7, cc = i & 127;
    float v = mid[(size_t)(n0 + rr) * 128 + cc];
    int k = cc & 31, kt = cc >> 5;
    sA[kt][rr + (((k >> 3) & 1) << 4)][(((k >> 4) & 1) << 3) | (k & 7)] =
        (_Float16)v;
  }
  __syncthreads();

  const int n = lane & 15;

  // GEMM1: [16,128]x[128,256]
  {
    v16h afr[4];
#pragma unroll
    for (int kt = 0; kt < 4; ++kt) afr[kt] = *(const v16h*)&sA[kt][lane][0];
    for (int nt = wave; nt < 16; nt += 4) {
      int col = nt * 16 + n;
      float bias = m1b[col];
      v8f c;
#pragma unroll
      for (int i = 0; i < 8; ++i) c[i] = bias;
#pragma unroll
      for (int kt = 0; kt < 4; ++kt) {
        v16h bfr = *(const v16h*)(m1wz + ((((kt * 16) + nt) * 32 + lane) << 4));
        c = __builtin_amdgcn_wmma_f32_16x16x32_f16(false, afr[kt], false, bfr,
                                                   (short)0, c, false, false);
      }
#pragma unroll
      for (int i = 0; i < 8; ++i) sT[i + ((lane >> 4) << 3)][col] = c[i];
    }
  }
  __syncthreads();

  if (tid < 16) {
    float s = 0.f, sq = 0.f;
    for (int cc = 0; cc < 256; ++cc) {
      float v = sT[tid][cc];
      s += v;
      sq += v * v;
    }
    float mu = s / 256.f;
    sMu[tid] = mu;
    sRs[tid] = rsqrtf(sq / 256.f - mu * mu + 1e-5f);
  }
  __syncthreads();
  for (int i = tid; i < 16 * 256; i += 128) {
    int rr = i >> 8, cc = i & 255;
    float v = (sT[rr][cc] - sMu[rr]) * sRs[rr] * lng[cc] + lnb[cc];
    v = v > 0.f ? v : 0.f;
    int k = cc & 31, kt = cc >> 5;
    sB[kt][rr + (((k >> 3) & 1) << 4)][(((k >> 4) & 1) << 3) | (k & 7)] =
        (_Float16)v;
  }
  __syncthreads();

  // GEMM2: [16,256]x[256,128]
  {
    v16h a2[8];
#pragma unroll
    for (int kt = 0; kt < 8; ++kt) a2[kt] = *(const v16h*)&sB[kt][lane][0];
    for (int nt = wave; nt < 8; nt += 4) {
      int col = nt * 16 + n;
      float bias = m2b[col];
      v8f c;
#pragma unroll
      for (int i = 0; i < 8; ++i) c[i] = bias;
#pragma unroll
      for (int kt = 0; kt < 8; ++kt) {
        v16h bfr = *(const v16h*)(m2wz + ((((kt * 8) + nt) * 32 + lane) << 4));
        c = __builtin_amdgcn_wmma_f32_16x16x32_f16(false, a2[kt], false, bfr,
                                                   (short)0, c, false, false);
      }
      // one base address, 8 stores/loads with immediate offsets
      int rr0 = (lane >> 4) << 3;
      float* op = hout + (size_t)(n0 + rr0) * 128 + col;
      const float* rp = hprev + (size_t)(n0 + rr0) * 128 + col;
#pragma unroll
      for (int i = 0; i < 8; ++i) {
        float v = c[i];
        if (residual) v += rp[(size_t)i * 128];
        op[(size_t)i * 128] = v;
      }
    }
  }
}

// ---------------------------------------------------------------------------
// Final linear: out = hin @ lin_w[128,112] + lin_b
// ---------------------------------------------------------------------------
__global__ void fishnet_out_kernel(const float* __restrict__ hin,
                                   const float* __restrict__ W,
                                   const float* __restrict__ b,
                                   float* __restrict__ out, int nNodes) {
  int idx = blockIdx.x * blockDim.x + threadIdx.x;
  int nn = idx / 112, j = idx % 112;
  if (nn >= nNodes) return;
  float acc = b[j];
  for (int k = 0; k < 128; ++k) acc += hin[(size_t)nn * 128 + k] * W[k * 112 + j];
  out[idx] = acc;
}

// ---------------------------------------------------------------------------
extern "C" void kernel_launch(void* const* d_in, const int* in_sizes, int n_in,
                              void* d_out, int out_size, void* d_ws,
                              size_t ws_size, hipStream_t stream) {
  const float* x      = (const float*)d_in[0];
  const float* eattr  = (const float*)d_in[1];
  const int*   eidx   = (const int*)d_in[2];
  const float* node_w = (const float*)d_in[3];
  const float* node_b = (const float*)d_in[4];
  const float* edge_w = (const float*)d_in[5];
  const float* edge_b = (const float*)d_in[6];
  const float* a1w    = (const float*)d_in[7];
  const float* a1b    = (const float*)d_in[8];
  const float* a2w    = (const float*)d_in[9];
  const float* a2b    = (const float*)d_in[10];
  const float* m1w    = (const float*)d_in[11];
  const float* m1b    = (const float*)d_in[12];
  const float* lng    = (const float*)d_in[13];
  const float* lnb    = (const float*)d_in[14];
  const float* m2w    = (const float*)d_in[15];
  const float* m2b    = (const float*)d_in[16];
  const float* norm_g = (const float*)d_in[17];
  const float* norm_b = (const float*)d_in[18];
  const float* lin_w  = (const float*)d_in[19];
  const float* lin_b  = (const float*)d_in[20];

  const int* srcIdx = eidx;
  const int* dstIdx = eidx + N_EDGES;

  const size_t NH = (size_t)N_NODES * HID;
  float* ws  = (float*)d_ws;
  float* h   = ws;
  float* hin = h + NH;
  float* mid = hin + NH;
  float* s_n = mid + NH;
  float* F_n = s_n + (size_t)N_NODES * NP;
  _Float16* a1wz = (_Float16*)(F_n + (size_t)N_NODES * NP * NP);
  _Float16* m1wz = a1wz + 3 * 4 * 10 * 512;   // [3][4][10][32][16]
  _Float16* m2wz = m1wz + 3 * 4 * 16 * 512;   // [3][4][16][32][16]
                                              // m2wz: [3][8][8][32][16]

  // weight pre-swizzle into WMMA-B fragment order (f16)
  fishnet_swz_kernel<<<(3 * 4 * 10 * 512 + 255) / 256, 256, 0, stream>>>(
      a1w, a1wz, 128, FDIM, 160, 3);
  fishnet_swz_kernel<<<(3 * 4 * 16 * 512 + 255) / 256, 256, 0, stream>>>(
      m1w, m1wz, 128, 256, 256, 3);
  fishnet_swz_kernel<<<(3 * 8 * 8 * 512 + 255) / 256, 256, 0, stream>>>(
      m2w, m2wz, 256, 128, 128, 3);

  fishnet_enc_kernel<<<(N_NODES * 128 + 127) / 128, 128, 0, stream>>>(
      x, node_w, node_b, h, N_NODES);

  for (int l = 0; l < 3; ++l) {
    const float* layer_in = h;
    if (l > 0) {
      fishnet_ln_kernel<<<N_NODES, 128, 0, stream>>>(
          h, norm_g + l * 128, norm_b + l * 128, hin, N_NODES, 1);
      layer_in = hin;
    }
    int nz4 = N_NODES * (NP + NP * NP) / 4;
    fishnet_zero_kernel<<<(nz4 + 255) / 256, 256, 0, stream>>>((float4*)s_n,
                                                               nz4);

    fishnet_edge_kernel<<<N_EDGES / 64, 128, 0, stream>>>(
        layer_in, eattr, srcIdx, dstIdx, edge_w, edge_b,
        a1wz + (size_t)l * 4 * 10 * 512, a1b + (size_t)l * FDIM, s_n, F_n);
    fishnet_solve_kernel<<<N_NODES / 4, 128, 0, stream>>>(
        F_n, s_n, a2w + (size_t)l * 16 * 128, a2b + (size_t)l * 128, layer_in,
        mid, N_NODES);
    fishnet_mlp_kernel<<<N_NODES / 16, 128, 0, stream>>>(
        mid, h, m1wz + (size_t)l * 4 * 16 * 512, m1b + (size_t)l * 256,
        lng + (size_t)l * 256, lnb + (size_t)l * 256,
        m2wz + (size_t)l * 8 * 8 * 512, m2b + (size_t)l * 128, h,
        (l > 0) ? 1 : 0, N_NODES);
  }

  fishnet_ln_kernel<<<N_NODES, 128, 0, stream>>>(h, norm_g, norm_b, hin,
                                                 N_NODES, 1);
  fishnet_out_kernel<<<(N_NODES * 112 + 127) / 128, 128, 0, stream>>>(
      hin, lin_w, lin_b, (float*)d_out, N_NODES);
}
